// Diffusion_ViTCA_NCA_27410481283283
// MI455X (gfx1250) — matrix operand order repaired and checked by reading, not compiled, not run on hardware
//
#include <hip/hip_runtime.h>
#include <math.h>

// ---------------- problem constants (from reference) ----------------
#define B_    8
#define H_    64
#define W_    64
#define C_    64
#define HID_  128
#define NTOK  (B_*H_*W_)      // 32768 tokens / spatial rows
#define STEPS_ 6

typedef __bf16 bf16_t;
typedef __attribute__((ext_vector_type(16))) __bf16 v16bf;
typedef __attribute__((ext_vector_type(8)))  __bf16 v8bf;
typedef __attribute__((ext_vector_type(8)))  float  v8f;

// fp32 -> bf16 round-to-nearest-even
__device__ __forceinline__ bf16_t f2bf(float f) {
  union { float f; unsigned u; } v; v.f = f;
  unsigned r = v.u + 0x7FFFu + ((v.u >> 16) & 1u);
  unsigned short h = (unsigned short)(r >> 16);
  bf16_t o; __builtin_memcpy(&o, &h, 2);
  return o;
}

// A fragment (16-bit A 16x32 layout, wave32):
//   lanes 0-15 : M=lane,     halves 0..7 = K0..7,   halves 8..15 = K16..23
//   lanes 16-31: M=lane-16,  halves 0..7 = K8..15,  halves 8..15 = K24..31
// hi = (lane>=16) ? 8 : 0
__device__ __forceinline__ v16bf load_afrag(const bf16_t* __restrict__ row, int k0, int hi) {
  union { v16bf v; v8bf h[2]; } u;
  u.h[0] = *(const v8bf*)(row + k0 + hi);
  u.h[1] = *(const v8bf*)(row + k0 + 16 + hi);
  return u.v;
}

// B fragment (16-bit B 32x16): lane L holds column N=n0+L%16; halves 0..7 = K hi..hi+7, halves 8..15 = K 16+hi..16+hi+7 (hi = L>=16 ? 8 : 0).
// Bm is [K][N] row-major bf16.
__device__ __forceinline__ v16bf load_bfrag(const bf16_t* __restrict__ Bm, int N,
                                            int k0, int n0, int lane) {
  int hi = (lane >= 16) ? 8 : 0;
  const bf16_t* p = Bm + (size_t)k0 * N + n0 + (lane & 15);
  union { v16bf v; bf16_t e[16]; } u;
  for (int s = 0; s < 8; ++s) {
    u.e[s] = p[(size_t)(hi + s) * N];
    u.e[8 + s] = p[(size_t)(16 + hi + s) * N];
  }
  return u.v;
}

#define WMMA_BF16(a, b, c) \
  __builtin_amdgcn_wmma_f32_16x16x32_bf16(false, (a), false, (b), (short)0, (c), false, false)

// ---------------- weight prep: f32 [N][K] -> bf16 [K][N] ----------------
__global__ void prep_transpose_kernel(const float* __restrict__ src, bf16_t* __restrict__ dst,
                                      int Nn, int Kk) {
  int i = blockIdx.x * blockDim.x + threadIdx.x;
  if (i >= Nn * Kk) return;
  int n = i / Kk, k = i - n * Kk;
  dst[(size_t)k * Nn + n] = f2bf(src[i]);
}

// conv weights: p0_w/p1_w [O=64][I=64][3][3] -> Wc [K=576][N=128], K=tap*64+ci, N=o (p0) / 64+o (p1)
__global__ void prep_conv_kernel(const float* __restrict__ w0, const float* __restrict__ w1,
                                 bf16_t* __restrict__ dst) {
  int i = blockIdx.x * blockDim.x + threadIdx.x;
  if (i >= 576 * 128) return;
  int k = i >> 7, n = i & 127;
  int tap = k >> 6, ci = k & 63;
  float v = (n < 64) ? w0[(size_t)(n * 64 + ci) * 9 + tap]
                     : w1[(size_t)((n - 64) * 64 + ci) * 9 + tap];
  dst[i] = f2bf(v);
}

// ---------------- per-step kernels ----------------

// P[:, 0:64] = bf16(X)   (P is [NTOK][192] bf16)
__global__ void cvt_x_kernel(const float* __restrict__ X, bf16_t* __restrict__ P) {
  int idx = blockIdx.x * blockDim.x + threadIdx.x;   // NTOK*64
  int m = idx >> 6, c = idx & 63;
  P[(size_t)m * 192 + c] = f2bf(X[idx]);
}

__global__ void zero_stats_kernel(float* __restrict__ stats) {
  if (threadIdx.x < 16) stats[threadIdx.x] = 0.f;
}

// conv-as-GEMM with on-the-fly im2col (reflect padding).
// 8 waves/block; wave -> (M-tile = blk*4 + wv>>1, N-half = wv&1).
// Each wave computes 16x64: A fragment loaded once, fed to 4 WMMAs.
__global__ void conv_gemm_kernel(const bf16_t* __restrict__ P, const bf16_t* __restrict__ Wc,
                                 const float* __restrict__ b0, const float* __restrict__ b1,
                                 bf16_t* __restrict__ Pout) {
  int lane = threadIdx.x & 31;
  int wv   = threadIdx.x >> 5;
  int mt   = blockIdx.x * 4 + (wv >> 1);
  int n0b_ = (wv & 1) * 64;
  int hi   = (lane >= 16) ? 8 : 0;
  int mrow = mt * 16 + (lane & 15);
  int b  = mrow >> 12;
  int hw = mrow & 4095;
  int h  = hw >> 6, w = hw & 63;
  v8f acc0 = {0.f,0.f,0.f,0.f,0.f,0.f,0.f,0.f};
  v8f acc1 = acc0, acc2 = acc0, acc3 = acc0;
  for (int tap = 0; tap < 9; ++tap) {
    int dy = tap / 3 - 1, dx = tap % 3 - 1;
    int rh = h + dy; rh = (rh < 0) ? 1 : ((rh > 63) ? 62 : rh);   // reflect pad=1
    int rw = w + dx; rw = (rw < 0) ? 1 : ((rw > 63) ? 62 : rw);
    const bf16_t* row = P + (size_t)((b * 64 + rh) * 64 + rw) * 192;
    for (int kc = 0; kc < 64; kc += 32) {
      v16bf a = load_afrag(row, kc, hi);
      int kk = tap * 64 + kc;
      acc0 = WMMA_BF16(a, load_bfrag(Wc, 128, kk, n0b_ +  0, lane), acc0);
      acc1 = WMMA_BF16(a, load_bfrag(Wc, 128, kk, n0b_ + 16, lane), acc1);
      acc2 = WMMA_BF16(a, load_bfrag(Wc, 128, kk, n0b_ + 32, lane), acc2);
      acc3 = WMMA_BF16(a, load_bfrag(Wc, 128, kk, n0b_ + 48, lane), acc3);
    }
  }
  int col = lane & 15;
  v8f accs[4] = {acc0, acc1, acc2, acc3};
  for (int j = 0; j < 4; ++j) {
    int n = n0b_ + j * 16 + col;
    float bias = (n < 64) ? b0[n] : b1[n - 64];
    for (int r = 0; r < 8; ++r) {
      int m = mt * 16 + r + hi;
      Pout[(size_t)m * 192 + 64 + n] = f2bf(accs[j][r] + bias);
    }
  }
}

// fc0: HD = relu(P @ W0^T + b), plus per-batch sum/sumsq for global LN.
// 8 waves/block; wave -> (M-tile = blk*4 + wv>>1, N-half = wv&1), 16x64 per wave.
__global__ void fc0_gemm_kernel(const bf16_t* __restrict__ P, const bf16_t* __restrict__ W0,
                                const float* __restrict__ bias, float* __restrict__ HD,
                                float* __restrict__ stats) {
  __shared__ float ls[256];
  __shared__ float lss[256];
  int lane = threadIdx.x & 31;
  int wv   = threadIdx.x >> 5;
  int mt   = blockIdx.x * 4 + (wv >> 1);
  int n0b_ = (wv & 1) * 64;
  int hi   = (lane >= 16) ? 8 : 0;
  const bf16_t* row = P + (size_t)(mt * 16 + (lane & 15)) * 192;
  v8f acc0 = {0.f,0.f,0.f,0.f,0.f,0.f,0.f,0.f};
  v8f acc1 = acc0, acc2 = acc0, acc3 = acc0;
  for (int k0 = 0; k0 < 192; k0 += 32) {
    v16bf a = load_afrag(row, k0, hi);
    acc0 = WMMA_BF16(a, load_bfrag(W0, 128, k0, n0b_ +  0, lane), acc0);
    acc1 = WMMA_BF16(a, load_bfrag(W0, 128, k0, n0b_ + 16, lane), acc1);
    acc2 = WMMA_BF16(a, load_bfrag(W0, 128, k0, n0b_ + 32, lane), acc2);
    acc3 = WMMA_BF16(a, load_bfrag(W0, 128, k0, n0b_ + 48, lane), acc3);
  }
  int col = lane & 15;
  v8f accs[4] = {acc0, acc1, acc2, acc3};
  float s = 0.f, ss = 0.f;
  for (int j = 0; j < 4; ++j) {
    int n = n0b_ + j * 16 + col;
    float bs = bias[n];
    for (int r = 0; r < 8; ++r) {
      int m = mt * 16 + r + hi;
      float v = fmaxf(accs[j][r] + bs, 0.f);
      HD[(size_t)m * 128 + n] = v;
      s += v; ss += v * v;
    }
  }
  ls[threadIdx.x] = s; lss[threadIdx.x] = ss;
  __syncthreads();
  for (int st = 128; st > 0; st >>= 1) {
    if (threadIdx.x < st) { ls[threadIdx.x] += ls[threadIdx.x + st]; lss[threadIdx.x] += lss[threadIdx.x + st]; }
    __syncthreads();
  }
  if (threadIdx.x == 0) {
    int bb = (blockIdx.x * 64) >> 12;   // 64 rows/block, all in one batch sample
    atomicAdd(&stats[bb * 2 + 0], ls[0]);
    atomicAdd(&stats[bb * 2 + 1], lss[0]);
  }
}

__global__ void stats_finalize_kernel(float* __restrict__ stats) {
  int b = threadIdx.x;
  if (b < 8) {
    const float Ninv = 1.0f / (float)(H_ * W_ * HID_);   // 524288
    float mean = stats[b * 2 + 0] * Ninv;
    float var  = stats[b * 2 + 1] * Ninv - mean * mean;
    stats[b * 2 + 0] = mean;
    stats[b * 2 + 1] = rsqrtf(var + 1e-5f);
  }
}

// fc1 with LayerNorm fused into the A-fragment build:
//   XN = X + mask * (norm0(HD) @ W1^T).
// 8 waves/block; one wave per M-tile, full N=64 (4 accumulators) so the
// expensive LN'd bf16 A-fragment is built exactly once per K-slice.
__global__ void fc1_gemm_kernel(const float* __restrict__ HD, const float* __restrict__ n0w,
                                const float* __restrict__ n0b, const float* __restrict__ stats,
                                const bf16_t* __restrict__ W1, const float* __restrict__ X,
                                const unsigned char* __restrict__ mask, float* __restrict__ XN) {
  int lane = threadIdx.x & 31;
  int wv   = threadIdx.x >> 5;
  int mt   = blockIdx.x * 8 + wv;
  int hi   = (lane >= 16) ? 8 : 0;
  int mrow = mt * 16 + (lane & 15);
  int bb   = mrow >> 12;
  int hw   = mrow & 4095;
  float mean = stats[bb * 2 + 0];
  float rstd = stats[bb * 2 + 1];
  const float* hrow = HD  + (size_t)mrow * 128;
  const float* wrow = n0w + (size_t)hw * 128;
  const float* brow = n0b + (size_t)hw * 128;
  v8f acc0 = {0.f,0.f,0.f,0.f,0.f,0.f,0.f,0.f};
  v8f acc1 = acc0, acc2 = acc0, acc3 = acc0;
  for (int k0 = 0; k0 < 128; k0 += 32) {
    union { v16bf v; bf16_t e[16]; } u;
    for (int j = 0; j < 8; ++j) {
      int k = k0 + hi + j;
      u.e[j] = f2bf((hrow[k] - mean) * rstd * wrow[k] + brow[k]);
    }
    for (int j = 0; j < 8; ++j) {
      int k = k0 + 16 + hi + j;
      u.e[8 + j] = f2bf((hrow[k] - mean) * rstd * wrow[k] + brow[k]);
    }
    acc0 = WMMA_BF16(u.v, load_bfrag(W1, 64, k0,  0, lane), acc0);
    acc1 = WMMA_BF16(u.v, load_bfrag(W1, 64, k0, 16, lane), acc1);
    acc2 = WMMA_BF16(u.v, load_bfrag(W1, 64, k0, 32, lane), acc2);
    acc3 = WMMA_BF16(u.v, load_bfrag(W1, 64, k0, 48, lane), acc3);
  }
  int col = lane & 15;
  v8f accs[4] = {acc0, acc1, acc2, acc3};
  for (int r = 0; r < 8; ++r) {
    int m = mt * 16 + r + hi;
    float mk = mask[m] ? 1.f : 0.f;
    for (int j = 0; j < 4; ++j) {
      int n = j * 16 + col;
      XN[(size_t)m * 64 + n] = X[(size_t)m * 64 + n] + accs[j][r] * mk;
    }
  }
}

// per-token LayerNorm over C=64 -> bf16
__global__ void ln_token_kernel(const float* __restrict__ T, const float* __restrict__ w,
                                const float* __restrict__ b, bf16_t* __restrict__ Y) {
  int m = blockIdx.x * blockDim.x + threadIdx.x;   // NTOK threads
  const float* r = T + (size_t)m * 64;
  float s = 0.f;
  for (int c = 0; c < 64; ++c) s += r[c];
  float mu = s * (1.0f / 64.0f);
  float var = 0.f;
  for (int c = 0; c < 64; ++c) { float d = r[c] - mu; var += d * d; }
  float rstd = rsqrtf(var * (1.0f / 64.0f) + 1e-5f);
  bf16_t* o = Y + (size_t)m * 64;
  for (int c = 0; c < 64; ++c) o[c] = f2bf((r[c] - mu) * rstd * w[c] + b[c]);
}

// generic WMMA GEMM, 16x64 per wave (4 accumulators, A fragment reused 4x).
// ngroups = N/64. 8 waves/block; wave gid -> (mtile, ngroup).
// act: 0 = none, 1 = exact GELU. res/outF/outB optional (nullptr to skip).
__global__ void gemm_epi_kernel(const bf16_t* __restrict__ A, int lda,
                                const bf16_t* __restrict__ Bm, int N, int K,
                                const float* __restrict__ bias, const float* __restrict__ res,
                                float* __restrict__ outF, bf16_t* __restrict__ outB,
                                int act, int ngroups) {
  int lane = threadIdx.x & 31;
  int gid  = blockIdx.x * (blockDim.x >> 5) + (threadIdx.x >> 5);
  int mt = gid / ngroups, g = gid - mt * ngroups;
  int n0b_ = g * 64;
  int hi = (lane >= 16) ? 8 : 0;
  const bf16_t* row = A + (size_t)(mt * 16 + (lane & 15)) * lda;
  v8f acc0 = {0.f,0.f,0.f,0.f,0.f,0.f,0.f,0.f};
  v8f acc1 = acc0, acc2 = acc0, acc3 = acc0;
  for (int k0 = 0; k0 < K; k0 += 32) {
    v16bf a = load_afrag(row, k0, hi);
    acc0 = WMMA_BF16(a, load_bfrag(Bm, N, k0, n0b_ +  0, lane), acc0);
    acc1 = WMMA_BF16(a, load_bfrag(Bm, N, k0, n0b_ + 16, lane), acc1);
    acc2 = WMMA_BF16(a, load_bfrag(Bm, N, k0, n0b_ + 32, lane), acc2);
    acc3 = WMMA_BF16(a, load_bfrag(Bm, N, k0, n0b_ + 48, lane), acc3);
  }
  int col = lane & 15;
  v8f accs[4] = {acc0, acc1, acc2, acc3};
  for (int j = 0; j < 4; ++j) {
    int n = n0b_ + j * 16 + col;
    float bs = bias ? bias[n] : 0.f;
    for (int r = 0; r < 8; ++r) {
      int m = mt * 16 + r + hi;
      float v = accs[j][r] + bs;
      if (act == 1) v = 0.5f * v * (1.f + erff(v * 0.70710678118f));
      if (res)  v += res[(size_t)m * N + n];
      if (outF) outF[(size_t)m * N + n] = v;
      if (outB) outB[(size_t)m * N + n] = f2bf(v);
    }
  }
}

// local 3x3-window softmax attention; zero-padded neighbourhoods.
// one thread per (token, head): QKV [NTOK][192] f32 -> O [NTOK][64] bf16
__global__ void attn_kernel(const float* __restrict__ QKV, bf16_t* __restrict__ O) {
  int t = blockIdx.x * blockDim.x + threadIdx.x;   // NTOK*4
  int head = t & 3, m = t >> 2;
  int b = m >> 12, h = (m >> 6) & 63, w = m & 63;
  const float4* q4 = (const float4*)(QKV + (size_t)m * 192 + head * 16);
  float4 qa = q4[0], qb = q4[1], qc = q4[2], qd = q4[3];
  float dots[9];
  const float scale = 0.25f;   // HD^-0.5 = 16^-0.5
  for (int tap = 0; tap < 9; ++tap) {
    int ry = h + tap / 3 - 1, rx = w + tap % 3 - 1;
    float d = 0.f;
    if (ry >= 0 && ry < 64 && rx >= 0 && rx < 64) {
      const float4* k4 = (const float4*)(QKV + (size_t)((b * 64 + ry) * 64 + rx) * 192 + 64 + head * 16);
      float4 ka = k4[0], kb = k4[1], kc = k4[2], kd = k4[3];
      d  = qa.x*ka.x + qa.y*ka.y + qa.z*ka.z + qa.w*ka.w;
      d += qb.x*kb.x + qb.y*kb.y + qb.z*kb.z + qb.w*kb.w;
      d += qc.x*kc.x + qc.y*kc.y + qc.z*kc.z + qc.w*kc.w;
      d += qd.x*kd.x + qd.y*kd.y + qd.z*kd.z + qd.w*kd.w;
    }
    dots[tap] = d * scale;     // OOB stays 0 (zero-padded keys)
  }
  float mx = dots[0];
  for (int tap = 1; tap < 9; ++tap) mx = fmaxf(mx, dots[tap]);
  float e[9], se = 0.f;
  for (int tap = 0; tap < 9; ++tap) { e[tap] = expf(dots[tap] - mx); se += e[tap]; }
  float inv = 1.f / se;
  float4 o0 = {0,0,0,0}, o1 = {0,0,0,0}, o2 = {0,0,0,0}, o3 = {0,0,0,0};
  for (int tap = 0; tap < 9; ++tap) {
    int ry = h + tap / 3 - 1, rx = w + tap % 3 - 1;
    if (ry >= 0 && ry < 64 && rx >= 0 && rx < 64) {
      const float4* v4 = (const float4*)(QKV + (size_t)((b * 64 + ry) * 64 + rx) * 192 + 128 + head * 16);
      float4 va = v4[0], vb = v4[1], vc = v4[2], vd = v4[3];
      float a = e[tap] * inv;
      o0.x += a*va.x; o0.y += a*va.y; o0.z += a*va.z; o0.w += a*va.w;
      o1.x += a*vb.x; o1.y += a*vb.y; o1.z += a*vb.z; o1.w += a*vb.w;
      o2.x += a*vc.x; o2.y += a*vc.y; o2.z += a*vc.z; o2.w += a*vc.w;
      o3.x += a*vd.x; o3.y += a*vd.y; o3.z += a*vd.z; o3.w += a*vd.w;
    }
  }
  bf16_t* op = O + (size_t)m * 64 + head * 16;
  float of[16] = {o0.x,o0.y,o0.z,o0.w, o1.x,o1.y,o1.z,o1.w,
                  o2.x,o2.y,o2.z,o2.w, o3.x,o3.y,o3.z,o3.w};
  for (int i = 0; i < 16; ++i) op[i] = f2bf(of[i]);
}

// X[:, 1:] = T3[:, 1:]   (channel 0 kept from pre-update state; in-place safe)
__global__ void finalize_kernel(const float* __restrict__ T3, float* __restrict__ X) {
  int idx = blockIdx.x * blockDim.x + threadIdx.x;   // NTOK*64
  if ((idx & 63) != 0) X[idx] = T3[idx];
}

// ---------------- host launch ----------------
extern "C" void kernel_launch(void* const* d_in, const int* in_sizes, int n_in,
                              void* d_out, int out_size, void* d_ws, size_t ws_size,
                              hipStream_t stream) {
  (void)in_sizes; (void)n_in; (void)out_size; (void)ws_size;

  const float* in_x     = (const float*)d_in[0];
  const unsigned char* in_masks = (const unsigned char*)d_in[1];   // bool array
  const float* p0_w  = (const float*)d_in[2];
  const float* p0_b  = (const float*)d_in[3];
  const float* p1_w  = (const float*)d_in[4];
  const float* p1_b  = (const float*)d_in[5];
  const float* fc0_w = (const float*)d_in[6];
  const float* fc0_b = (const float*)d_in[7];
  const float* fc1_w = (const float*)d_in[8];
  const float* n0w   = (const float*)d_in[9];
  const float* n0b   = (const float*)d_in[10];
  const float* ln1_w = (const float*)d_in[11];
  const float* ln1_b = (const float*)d_in[12];
  const float* qkv_w = (const float*)d_in[13];
  const float* out_w = (const float*)d_in[14];
  const float* out_b = (const float*)d_in[15];
  const float* ln2_w = (const float*)d_in[16];
  const float* ln2_b = (const float*)d_in[17];
  const float* ff1_w = (const float*)d_in[18];
  const float* ff1_b = (const float*)d_in[19];
  const float* ff2_w = (const float*)d_in[20];
  const float* ff2_b = (const float*)d_in[21];

  char* ws = (char*)d_ws;
  size_t off = 0;
  auto alloc = [&](size_t bytes) { char* p = ws + off; off += (bytes + 255) & ~(size_t)255; return p; };

  float*  X     = (float*)  alloc((size_t)NTOK * 64 * 4);
  bf16_t* P     = (bf16_t*) alloc((size_t)NTOK * 192 * 2);
  float*  HD    = (float*)  alloc((size_t)NTOK * 128 * 4);
  float*  XN    = (float*)  alloc((size_t)NTOK * 64 * 4);
  bf16_t* YB    = (bf16_t*) alloc((size_t)NTOK * 64 * 2);
  float*  QKV   = (float*)  alloc((size_t)NTOK * 192 * 4);
  bf16_t* O     = (bf16_t*) alloc((size_t)NTOK * 64 * 2);
  float*  T2    = (float*)  alloc((size_t)NTOK * 64 * 4);
  bf16_t* G     = (bf16_t*) alloc((size_t)NTOK * 64 * 2);
  float*  T3    = (float*)  alloc((size_t)NTOK * 64 * 4);
  float*  STATS = (float*)  alloc(256);
  bf16_t* WCONV = (bf16_t*) alloc((size_t)576 * 128 * 2);
  bf16_t* W0T   = (bf16_t*) alloc((size_t)192 * 128 * 2);
  bf16_t* W1T   = (bf16_t*) alloc((size_t)128 * 64 * 2);
  bf16_t* WQKVT = (bf16_t*) alloc((size_t)64 * 192 * 2);
  bf16_t* WOUTT = (bf16_t*) alloc((size_t)64 * 64 * 2);
  bf16_t* WFF1T = (bf16_t*) alloc((size_t)64 * 64 * 2);
  bf16_t* WFF2T = (bf16_t*) alloc((size_t)64 * 64 * 2);

  // initial state
  hipMemcpyAsync(X, in_x, (size_t)NTOK * 64 * 4, hipMemcpyDeviceToDevice, stream);

  // weight prep (bf16, [K][N])
  prep_conv_kernel<<<(576 * 128 + 255) / 256, 256, 0, stream>>>(p0_w, p1_w, WCONV);
  prep_transpose_kernel<<<(128 * 192 + 255) / 256, 256, 0, stream>>>(fc0_w, W0T, 128, 192);
  prep_transpose_kernel<<<(64 * 128 + 255) / 256, 256, 0, stream>>>(fc1_w, W1T, 64, 128);
  prep_transpose_kernel<<<(192 * 64 + 255) / 256, 256, 0, stream>>>(qkv_w, WQKVT, 192, 64);
  prep_transpose_kernel<<<(64 * 64 + 255) / 256, 256, 0, stream>>>(out_w, WOUTT, 64, 64);
  prep_transpose_kernel<<<(64 * 64 + 255) / 256, 256, 0, stream>>>(ff1_w, WFF1T, 64, 64);
  prep_transpose_kernel<<<(64 * 64 + 255) / 256, 256, 0, stream>>>(ff2_w, WFF2T, 64, 64);

  const int MT = NTOK / 16;   // 2048 M-tiles

  for (int s = 0; s < STEPS_; ++s) {
    const unsigned char* mask_s = in_masks + (size_t)s * NTOK;

    zero_stats_kernel<<<1, 64, 0, stream>>>(STATS);
    cvt_x_kernel<<<(NTOK * 64) / 256, 256, 0, stream>>>(X, P);

    // perception convs (y1,y2) -> P[:,64:192]; 4 M-tiles x 2 N-halves per block
    conv_gemm_kernel<<<MT / 4, 256, 0, stream>>>(P, WCONV, p0_b, p1_b, P);

    // fc0 + relu + LN stats
    fc0_gemm_kernel<<<MT / 4, 256, 0, stream>>>(P, W0T, fc0_b, HD, STATS);
    stats_finalize_kernel<<<1, 32, 0, stream>>>(STATS);

    // norm0 + fc1 + stochastic mask + residual -> XN; one M-tile per wave
    fc1_gemm_kernel<<<MT / 8, 256, 0, stream>>>(HD, n0w, n0b, STATS, W1T, X, mask_s, XN);

    // attention block
    ln_token_kernel<<<NTOK / 256, 256, 0, stream>>>(XN, ln1_w, ln1_b, YB);
    gemm_epi_kernel<<<MT * 3 / 8, 256, 0, stream>>>(YB, 64, WQKVT, 192, 64,
                                                    nullptr, nullptr, QKV, nullptr, 0, 3);
    attn_kernel<<<(NTOK * 4) / 256, 256, 0, stream>>>(QKV, O);
    gemm_epi_kernel<<<MT / 8, 256, 0, stream>>>(O, 64, WOUTT, 64, 64,
                                                out_b, XN, T2, nullptr, 0, 1);

    // feed-forward block
    ln_token_kernel<<<NTOK / 256, 256, 0, stream>>>(T2, ln2_w, ln2_b, YB);
    gemm_epi_kernel<<<MT / 8, 256, 0, stream>>>(YB, 64, WFF1T, 64, 64,
                                                ff1_b, nullptr, nullptr, G, 1, 1);
    gemm_epi_kernel<<<MT / 8, 256, 0, stream>>>(G, 64, WFF2T, 64, 64,
                                                ff2_b, T2, T3, nullptr, 0, 1);

    // keep input channel 0, update the rest (in-place into X)
    finalize_kernel<<<(NTOK * 64) / 256, 256, 0, stream>>>(T3, X);
  }

  hipMemcpyAsync(d_out, X, (size_t)NTOK * 64 * 4, hipMemcpyDeviceToDevice, stream);
}
